// kMeanCluster_30829275251214
// MI455X (gfx1250) — compile-verified, matrix operand order checked by reading, not run on hardware
//
#include <hip/hip_runtime.h>

typedef __attribute__((ext_vector_type(16))) __bf16 v16bf;
typedef __attribute__((ext_vector_type(8)))  __bf16 v8bf;
typedef __attribute__((ext_vector_type(8)))  float  v8f;

#define D_DIM 256
#define K_CL  512
#define MT 2   // 16-row M tiles per wave
#define NT 4   // 16-col N tiles per wave

// ---------- helpers ----------
__device__ __forceinline__ unsigned int bf16_rne_bits(float x) {
    unsigned int u = __builtin_bit_cast(unsigned int, x);
    return (u + 0x7FFFu + ((u >> 16) & 1u)) >> 16;
}

__device__ __forceinline__ v16bf make_frag(v8bf c0, v8bf c1) {
    union { struct { v8bf a, b; } s; v16bf v; } u;
    u.s.a = c0; u.s.b = c1;
    return u.v;
}

// ---------- pre-pass: f32 -> (bf16 hi, bf16 lo) split + row norms ----------
// One wave per 256-element row; lane handles 8 contiguous elements.
__global__ __launch_bounds__(256) void convert_split_kernel(
    const float* __restrict__ F, const float* __restrict__ C,
    unsigned short* __restrict__ Fhi, unsigned short* __restrict__ Flo,
    unsigned short* __restrict__ Chi, unsigned short* __restrict__ Clo,
    float* __restrict__ f2, float* __restrict__ c2, int M)
{
    const int lane = threadIdx.x & 31;
    const int wid  = threadIdx.x >> 5;
    const int row  = blockIdx.x * 8 + wid;

    const float* src; unsigned short *dhi, *dlo; float* nrm;
    if (row < M) {
        src = F + (size_t)row * D_DIM;
        dhi = Fhi + (size_t)row * D_DIM; dlo = Flo + (size_t)row * D_DIM;
        nrm = f2 + row;
    } else if (row < M + K_CL) {
        int r = row - M;
        src = C + (size_t)r * D_DIM;
        dhi = Chi + (size_t)r * D_DIM; dlo = Clo + (size_t)r * D_DIM;
        nrm = c2 + r;
    } else {
        return;
    }

    const float4* s4 = (const float4*)src + lane * 2;
    float4 a = s4[0], b = s4[1];
    float v[8] = {a.x, a.y, a.z, a.w, b.x, b.y, b.z, b.w};

    unsigned int hp[4], lp[4];
    float ss = 0.0f;
#pragma unroll
    for (int i = 0; i < 4; ++i) {
        float x0 = v[2*i], x1 = v[2*i+1];
        unsigned int h0 = bf16_rne_bits(x0), h1 = bf16_rne_bits(x1);
        float r0 = x0 - __builtin_bit_cast(float, h0 << 16);   // exact residual
        float r1 = x1 - __builtin_bit_cast(float, h1 << 16);
        unsigned int l0 = bf16_rne_bits(r0), l1 = bf16_rne_bits(r1);
        hp[i] = h0 | (h1 << 16);
        lp[i] = l0 | (l1 << 16);
        ss += x0 * x0 + x1 * x1;
    }
    *(uint4*)(dhi + lane * 8) = make_uint4(hp[0], hp[1], hp[2], hp[3]);
    *(uint4*)(dlo + lane * 8) = make_uint4(lp[0], lp[1], lp[2], lp[3]);

#pragma unroll
    for (int off = 16; off > 0; off >>= 1) ss += __shfl_xor(ss, off, 32);
    if (lane == 0) *nrm = ss;
}

// ---------- main GEMM: dist = f2 + c2 - 2 * (A_hi+A_lo)(C_hi+C_lo)^T ----------
// Wave tile: 32 rows x 64 cols. 8 waves/block cover all 512 cols for the
// block's 32 feature rows (feature reuse out of WGP$; centers L2-resident).
__global__ __launch_bounds__(256) void dist_wmma_kernel(
    const unsigned short* __restrict__ Fhi, const unsigned short* __restrict__ Flo,
    const unsigned short* __restrict__ Chi, const unsigned short* __restrict__ Clo,
    const float* __restrict__ f2, const float* __restrict__ c2,
    float* __restrict__ out)
{
    const int lane  = threadIdx.x & 31;
    const int wid   = threadIdx.x >> 5;
    const int row   = lane & 15;
    const int half  = lane >> 4;
    const int mBase = blockIdx.x * (MT * 16);
    const int nBase = wid * (NT * 16);

    // Per-lane fragment base pointers (chunk pair at +0 / +2 v8bf units; step 4 per K=32)
    const v8bf* pfh[MT]; const v8bf* pfl[MT];
#pragma unroll
    for (int mt = 0; mt < MT; ++mt) {
        size_t r = (size_t)(mBase + mt * 16 + row) * D_DIM;
        pfh[mt] = (const v8bf*)(Fhi + r) + half;
        pfl[mt] = (const v8bf*)(Flo + r) + half;
    }
    const v8bf* pch[NT]; const v8bf* pcl[NT];
#pragma unroll
    for (int j = 0; j < NT; ++j) {
        size_t r = (size_t)(nBase + j * 16 + row) * D_DIM;
        pch[j] = (const v8bf*)(Chi + r) + half;
        pcl[j] = (const v8bf*)(Clo + r) + half;
    }

    v8f acc[MT][NT] = {};

#pragma unroll
    for (int kk = 0; kk < D_DIM / 32; ++kk) {
        const int o0 = kk * 4, o1 = kk * 4 + 2;
        v16bf ah[MT], al[MT], bh[NT], bl[NT];
#pragma unroll
        for (int mt = 0; mt < MT; ++mt) {
            ah[mt] = make_frag(pfh[mt][o0], pfh[mt][o1]);
            al[mt] = make_frag(pfl[mt][o0], pfl[mt][o1]);
        }
#pragma unroll
        for (int j = 0; j < NT; ++j) {
            bh[j] = make_frag(pch[j][o0], pch[j][o1]);
            bl[j] = make_frag(pcl[j][o0], pcl[j][o1]);
        }
#pragma unroll
        for (int mt = 0; mt < MT; ++mt) {
#pragma unroll
            for (int j = 0; j < NT; ++j) {
                acc[mt][j] = __builtin_amdgcn_wmma_f32_16x16x32_bf16(
                    false, ah[mt], false, bh[j], (short)0, acc[mt][j], false, false);
                acc[mt][j] = __builtin_amdgcn_wmma_f32_16x16x32_bf16(
                    false, ah[mt], false, bl[j], (short)0, acc[mt][j], false, false);
                acc[mt][j] = __builtin_amdgcn_wmma_f32_16x16x32_bf16(
                    false, al[mt], false, bh[j], (short)0, acc[mt][j], false, false);
            }
        }
    }

    // Epilogue: dist = f2[m] + c2[n] - 2*cross
    float f2v = f2[mBase + lane];                 // lane -> row mBase+lane (covers MT*16=32 rows)
    float c2v[NT];
#pragma unroll
    for (int j = 0; j < NT; ++j) c2v[j] = c2[nBase + j * 16 + row];

#pragma unroll
    for (int mt = 0; mt < MT; ++mt) {
#pragma unroll
        for (int r = 0; r < 8; ++r) {
            // C layout: VGPR r, lanes 0-15 -> M=r, N=lane; lanes 16-31 -> M=r+8, N=lane-16
            float fm = __shfl(f2v, mt * 16 + r + 8 * half, 32);
            float* orow = out + (size_t)(mBase + mt * 16 + r + 8 * half) * K_CL;
#pragma unroll
            for (int j = 0; j < NT; ++j) {
                orow[nBase + j * 16 + row] = fm + c2v[j] - 2.0f * acc[mt][j][r];
            }
        }
    }
}

extern "C" void kernel_launch(void* const* d_in, const int* in_sizes, int n_in,
                              void* d_out, int out_size, void* d_ws, size_t ws_size,
                              hipStream_t stream) {
    const float* F = (const float*)d_in[0];   // (B,S,D) = (8,4096,256)
    const float* C = (const float*)d_in[1];   // (1,K,D) = (1,512,256)
    float* out = (float*)d_out;               // (B,S,K) f32

    const int M = in_sizes[0] / D_DIM;        // 32768
    const size_t fE = (size_t)M * D_DIM;
    const size_t cE = (size_t)K_CL * D_DIM;

    unsigned char* ws = (unsigned char*)d_ws;
    unsigned short* Fhi = (unsigned short*)ws;            ws += fE * sizeof(unsigned short);
    unsigned short* Flo = (unsigned short*)ws;            ws += fE * sizeof(unsigned short);
    unsigned short* Chi = (unsigned short*)ws;            ws += cE * sizeof(unsigned short);
    unsigned short* Clo = (unsigned short*)ws;            ws += cE * sizeof(unsigned short);
    float* f2 = (float*)ws;                               ws += (size_t)M * sizeof(float);
    float* c2 = (float*)ws;

    // Pre-pass: split-bf16 conversion + exact f32 row norms (one wave per row).
    int rows = M + K_CL;
    convert_split_kernel<<<(rows + 7) / 8, 256, 0, stream>>>(
        F, C, Fhi, Flo, Chi, Clo, f2, c2, M);

    // Main GEMM: grid covers M/32 blocks x (8 waves -> full K=512).
    dist_wmma_kernel<<<M / (MT * 16), 256, 0, stream>>>(
        Fhi, Flo, Chi, Clo, f2, c2, out);
}
